// CodeRepairTransformer_40389872451603
// MI455X (gfx1250) — matrix operand order repaired and checked by reading, not compiled, not run on hardware
//
#include <hip/hip_runtime.h>

// ---------------------------------------------------------------------------
// BERT-style encoder-decoder forward for MI455X (gfx1250, wave32, WMMA).
// All dense GEMMs run on v_wmma_f32_16x16x32_f16 (f16 in, f32 accum),
// double-buffered through LDS with register prefetch (1 barrier / K-step).
// Staging uses b64 global loads + packed b32 LDS stores.
// ---------------------------------------------------------------------------

typedef __attribute__((ext_vector_type(16))) _Float16 v16h;
typedef __attribute__((ext_vector_type(8)))  float    v8f;

constexpr int kV  = 32000;
constexpr int kD  = 768;
constexpr int kH  = 12;
constexpr int kF  = 3072;
constexpr int kL  = 6;
constexpr int kB  = 8;
constexpr int kS  = 512;
constexpr int kHD = 64;
constexpr int kM  = kB * kS;      // 4096 rows for every GEMM

__device__ __forceinline__ unsigned pack_h2(float x, float y) {
    union { _Float16 h[2]; unsigned u; } p;
    p.h[0] = (_Float16)x;
    p.h[1] = (_Float16)y;
    return p.u;
}

// ---------------------------------------------------------------------------
// WMMA GEMM:  C[M,N] = act(A[M,K] @ W[K,N] + bias)
// Block tile 128(M) x 128(N), 8 waves (4x2), each wave 32x64 via 2x4 WMMA.
// K stepped by 32, double-buffered LDS + register prefetch.
// ACT: 0 = none, 1 = exact-erf GELU.
// ---------------------------------------------------------------------------
template <int ACT>
__global__ __launch_bounds__(256) void k_gemm(const float* __restrict__ A,
                                              const float* __restrict__ W,
                                              const float* __restrict__ bias,
                                              float* __restrict__ C,
                                              int N, int K) {
    __shared__ _Float16 As[2][128][34];   // [buf][m][k], +2 halves pad
    __shared__ _Float16 Bs[2][128][34];   // [buf][n][k] (W tile transposed)

    const int tid  = threadIdx.x;
    const int lane = tid & 31;
    const int wid  = tid >> 5;
    const int wm   = (wid >> 1) * 32;     // wave M offset: 0,32,64,96
    const int wn   = (wid & 1)  * 64;     // wave N offset: 0,64
    const int m16  = lane & 15;
    const int hi   = lane >> 4;

    const long bm = (long)blockIdx.y * 128;
    const long bn = (long)blockIdx.x * 128;

    // staging coordinates
    const int ak = (tid & 15) * 2;        // A: k-pair base 0..30
    const int am = tid >> 4;              // A: row 0..15 (x8 rows of 16)
    const int bc = tid & 127;             // B: n-col 0..127
    const int bg = (tid >> 7) * 2;        // B: k-pair base 0 or 2 (step 4)

    v8f acc[2][4] = {};
    float2 aR[8];
    float  bR0[8], bR1[8];

    // ---- prologue: stage K-slice 0 into buffer 0 ----------------------
    {
        const float* Ap = A + (bm + am) * (long)K + ak;
#pragma unroll
        for (int rr = 0; rr < 8; ++rr)
            aR[rr] = *(const float2*)(Ap + (long)rr * 16 * K);
#pragma unroll
        for (int p = 0; p < 8; ++p) {
            const long kr = bg + 4 * p;
            bR0[p] = W[kr * N + bn + bc];
            bR1[p] = W[(kr + 1) * N + bn + bc];
        }
#pragma unroll
        for (int rr = 0; rr < 8; ++rr)
            *reinterpret_cast<unsigned*>(&As[0][am + rr * 16][ak]) =
                pack_h2(aR[rr].x, aR[rr].y);
#pragma unroll
        for (int p = 0; p < 8; ++p)
            *reinterpret_cast<unsigned*>(&Bs[0][bc][bg + 4 * p]) =
                pack_h2(bR0[p], bR1[p]);
    }
    __syncthreads();

    int cur = 0;
    for (int k0 = 0; k0 < K; k0 += 32) {
        const bool nxt = (k0 + 32) < K;

        // ---- prefetch next K-slice into registers (hides HBM latency) --
        if (nxt) {
            const float* Ap = A + (bm + am) * (long)K + (k0 + 32) + ak;
#pragma unroll
            for (int rr = 0; rr < 8; ++rr)
                aR[rr] = *(const float2*)(Ap + (long)rr * 16 * K);
#pragma unroll
            for (int p = 0; p < 8; ++p) {
                const long kr = k0 + 32 + bg + 4 * p;
                bR0[p] = W[kr * N + bn + bc];
                bR1[p] = W[(kr + 1) * N + bn + bc];
            }
        }

        // ---- load fragments per ISA 16-bit A / B layouts ---------------
        v16h afrag[2], bfrag[4];
#pragma unroll
        for (int i = 0; i < 2; ++i) {
            const int row = wm + 16 * i + m16;
#pragma unroll
            for (int j = 0; j < 8; ++j) {
                const int kk = 16 * (j >> 2) + 8 * hi + 2 * (j & 3);
                afrag[i][2 * j]     = As[cur][row][kk];
                afrag[i][2 * j + 1] = As[cur][row][kk + 1];
            }
        }
#pragma unroll
        for (int j = 0; j < 4; ++j) {
            const int coln = wn + 16 * j + m16;
#pragma unroll
            for (int e = 0; e < 16; ++e)
                bfrag[j][e] = Bs[cur][coln][16 * (e >> 3) + 8 * hi + (e & 7)];
        }

        // ---- 2x4 WMMA, in-register f32 accumulation --------------------
#pragma unroll
        for (int i = 0; i < 2; ++i)
#pragma unroll
            for (int j = 0; j < 4; ++j)
                acc[i][j] = __builtin_amdgcn_wmma_f32_16x16x32_f16(
                    false, afrag[i], false, bfrag[j],
                    (short)0, acc[i][j], false, false);

        // ---- publish next K-slice into the alternate buffer ------------
        if (nxt) {
#pragma unroll
            for (int rr = 0; rr < 8; ++rr)
                *reinterpret_cast<unsigned*>(&As[cur ^ 1][am + rr * 16][ak]) =
                    pack_h2(aR[rr].x, aR[rr].y);
#pragma unroll
            for (int p = 0; p < 8; ++p)
                *reinterpret_cast<unsigned*>(&Bs[cur ^ 1][bc][bg + 4 * p]) =
                    pack_h2(bR0[p], bR1[p]);
        }
        __syncthreads();
        cur ^= 1;
    }

    // ---- epilogue: bias (+GELU), C/D layout row = v + 8*hi, col = m16 --
#pragma unroll
    for (int i = 0; i < 2; ++i)
#pragma unroll
        for (int j = 0; j < 4; ++j) {
            const long col = bn + wn + 16 * j + m16;
            const float bv = bias[col];
#pragma unroll
            for (int v = 0; v < 8; ++v) {
                const long row = bm + wm + 16 * i + v + 8 * hi;
                float x = acc[i][j][v] + bv;
                if (ACT == 1)
                    x = 0.5f * x * (1.0f + erff(x * 0.7071067811865475f));
                C[row * (long)N + col] = x;
            }
        }
}

// ---------------------------------------------------------------------------
// Embedding gather + LayerNorm.  One block per token row (768 wide).
// ---------------------------------------------------------------------------
__global__ __launch_bounds__(256) void k_embed_ln(const int* __restrict__ ids,
                                                  const float* __restrict__ tok,
                                                  const float* __restrict__ pos,
                                                  const float* __restrict__ lw,
                                                  const float* __restrict__ lb,
                                                  float* __restrict__ out) {
    const int row = blockIdx.x;           // b*S + s
    const int s   = row % kS;
    const long id = ids[row];
    __shared__ float xs[kD];
    __shared__ float rs[256], rq[256];

    float sum = 0.f, sq = 0.f;
#pragma unroll
    for (int t = 0; t < 3; ++t) {
        const int d = threadIdx.x + 256 * t;
        const float x = tok[id * kD + d] + pos[(long)s * kD + d];
        xs[d] = x; sum += x; sq += x * x;
    }
    rs[threadIdx.x] = sum; rq[threadIdx.x] = sq;
    __syncthreads();
    for (int st = 128; st > 0; st >>= 1) {
        if (threadIdx.x < st) {
            rs[threadIdx.x] += rs[threadIdx.x + st];
            rq[threadIdx.x] += rq[threadIdx.x + st];
        }
        __syncthreads();
    }
    const float mu  = rs[0] * (1.f / kD);
    const float var = rq[0] * (1.f / kD) - mu * mu;
    const float inv = rsqrtf(var + 1e-12f);
#pragma unroll
    for (int t = 0; t < 3; ++t) {
        const int d = threadIdx.x + 256 * t;
        out[(long)row * kD + d] = (xs[d] - mu) * inv * lw[d] + lb[d];
    }
}

// ---------------------------------------------------------------------------
// Residual add + LayerNorm (safe in-place: out may alias a).
// ---------------------------------------------------------------------------
__global__ __launch_bounds__(256) void k_add_ln(const float* __restrict__ a,
                                                const float* __restrict__ b,
                                                const float* __restrict__ lw,
                                                const float* __restrict__ lb,
                                                float* __restrict__ out) {
    const int row = blockIdx.x;
    __shared__ float xs[kD];
    __shared__ float rs[256], rq[256];

    float sum = 0.f, sq = 0.f;
#pragma unroll
    for (int t = 0; t < 3; ++t) {
        const int d = threadIdx.x + 256 * t;
        const float x = a[(long)row * kD + d] + b[(long)row * kD + d];
        xs[d] = x; sum += x; sq += x * x;
    }
    rs[threadIdx.x] = sum; rq[threadIdx.x] = sq;
    __syncthreads();
    for (int st = 128; st > 0; st >>= 1) {
        if (threadIdx.x < st) {
            rs[threadIdx.x] += rs[threadIdx.x + st];
            rq[threadIdx.x] += rq[threadIdx.x + st];
        }
        __syncthreads();
    }
    const float mu  = rs[0] * (1.f / kD);
    const float var = rq[0] * (1.f / kD) - mu * mu;
    const float inv = rsqrtf(var + 1e-12f);
#pragma unroll
    for (int t = 0; t < 3; ++t) {
        const int d = threadIdx.x + 256 * t;
        out[(long)row * kD + d] = (xs[d] - mu) * inv * lw[d] + lb[d];
    }
}

// ---------------------------------------------------------------------------
// Attention: one 64-thread block per (qi, h, b).  Two-pass streaming softmax.
// scores = q.k/8 + (1-mask)*(-1e4); ctx = softmax(scores) @ V.
// ---------------------------------------------------------------------------
__global__ __launch_bounds__(64) void k_attn(const float* __restrict__ q,
                                             const float* __restrict__ k,
                                             const float* __restrict__ v,
                                             const int* __restrict__ mask,
                                             float* __restrict__ ctx) {
    const int qi = blockIdx.x, h = blockIdx.y, b = blockIdx.z;
    const int tid = threadIdx.x;
    __shared__ float qs[kHD];
    __shared__ float sc[kS];
    __shared__ float red[64];

    qs[tid] = q[((long)(b * kS + qi)) * kD + h * kHD + tid];
    __syncthreads();

    // pass 1: scores
    for (int kk = tid; kk < kS; kk += 64) {
        const float* kp = k + ((long)(b * kS + kk)) * kD + h * kHD;
        float d0 = 0.f;
#pragma unroll
        for (int dd = 0; dd < kHD; ++dd) d0 += qs[dd] * kp[dd];
        const float mv = (float)mask[b * kS + kk];
        sc[kk] = d0 * 0.125f + (1.0f - mv) * (-10000.0f);
    }
    __syncthreads();

    // max
    float mx = -3.0e38f;
    for (int kk = tid; kk < kS; kk += 64) mx = fmaxf(mx, sc[kk]);
    red[tid] = mx;
    __syncthreads();
    for (int st = 32; st > 0; st >>= 1) {
        if (tid < st) red[tid] = fmaxf(red[tid], red[tid + st]);
        __syncthreads();
    }
    mx = red[0];
    __syncthreads();

    // exp + sum
    float sm = 0.f;
    for (int kk = tid; kk < kS; kk += 64) {
        const float e = expf(sc[kk] - mx);
        sc[kk] = e; sm += e;
    }
    red[tid] = sm;
    __syncthreads();
    for (int st = 32; st > 0; st >>= 1) {
        if (tid < st) red[tid] += red[tid + st];
        __syncthreads();
    }
    const float inv = 1.0f / red[0];
    __syncthreads();

    // pass 2: thread owns output dim `tid`
    float acc = 0.f;
    for (int kk = 0; kk < kS; ++kk)
        acc += sc[kk] * v[((long)(b * kS + kk)) * kD + h * kHD + tid];
    ctx[((long)(b * kS + qi)) * kD + h * kHD + tid] = acc * inv;
}

// ---------------------------------------------------------------------------
// Host-side orchestration
// ---------------------------------------------------------------------------
// d_in flattening (depth-first insertion order of setup_inputs dict):
enum {
    I_TOK = 0, I_POS, I_LNW, I_LNB,
    // encoder
    E_WQ, E_BQ, E_WK, E_BK, E_WV, E_BV, E_WO, E_BO,
    E_W1, E_B1, E_W2, E_B2, E_L1W, E_L1B, E_L2W, E_L2B,
    // decoder
    D_SWQ, D_SBQ, D_SWK, D_SBK, D_SWV, D_SBV, D_SWO, D_SBO,
    D_CWQ, D_CBQ, D_CWK, D_CBK, D_CWV, D_CBV, D_CWO, D_CBO,
    D_W1, D_B1, D_W2, D_B2,
    D_L1W, D_L1B, D_L2W, D_L2B, D_L3W, D_L3B,
    I_OUTW, I_OUTB,
    I_IDS, I_AMASK, I_LABELS, I_DMASK
};

static inline void launch_gemm(const float* A, const float* W, const float* bias,
                               float* C, int N, int K, int act, hipStream_t st) {
    dim3 grid(N / 128, kM / 128);
    if (act) k_gemm<1><<<grid, 256, 0, st>>>(A, W, bias, C, N, K);
    else     k_gemm<0><<<grid, 256, 0, st>>>(A, W, bias, C, N, K);
}

extern "C" void kernel_launch(void* const* d_in, const int* in_sizes, int n_in,
                              void* d_out, int out_size, void* d_ws, size_t ws_size,
                              hipStream_t stream) {
    (void)in_sizes; (void)n_in; (void)out_size; (void)ws_size;

    const float* F32[48];
    for (int i = 0; i < 48; ++i) F32[i] = (const float*)d_in[i];
    const int* ids    = (const int*)d_in[I_IDS];
    const int* amask  = (const int*)d_in[I_AMASK];
    const int* labels = (const int*)d_in[I_LABELS];
    const int* dmask  = (const int*)d_in[I_DMASK];
    float* logits = (float*)d_out;

    // workspace layout (fp32)
    const size_t SZ = (size_t)kM * kD;           // token-major activation
    float* ws    = (float*)d_ws;
    float* h_enc = ws;
    float* h_dec = h_enc + SZ;
    float* tq    = h_dec + SZ;
    float* tk    = tq + SZ;
    float* tv    = tk + SZ;
    float* tctx  = tv + SZ;
    float* tp    = tctx + SZ;
    float* tf    = tp + SZ;                      // [kM, kF]

    const dim3 attn_grid(kS, kH, kB);
    const int DD = kD * kD, DL = kD, FL = kF, DF = kD * kF;

    // ---------------- encoder ----------------
    k_embed_ln<<<kM, 256, 0, stream>>>(ids, F32[I_TOK], F32[I_POS],
                                       F32[I_LNW], F32[I_LNB], h_enc);
    for (int i = 0; i < kL; ++i) {
        launch_gemm(h_enc, F32[E_WQ] + (long)i * DD, F32[E_BQ] + (long)i * DL, tq, kD, kD, 0, stream);
        launch_gemm(h_enc, F32[E_WK] + (long)i * DD, F32[E_BK] + (long)i * DL, tk, kD, kD, 0, stream);
        launch_gemm(h_enc, F32[E_WV] + (long)i * DD, F32[E_BV] + (long)i * DL, tv, kD, kD, 0, stream);
        k_attn<<<attn_grid, 64, 0, stream>>>(tq, tk, tv, amask, tctx);
        launch_gemm(tctx, F32[E_WO] + (long)i * DD, F32[E_BO] + (long)i * DL, tp, kD, kD, 0, stream);
        k_add_ln<<<kM, 256, 0, stream>>>(h_enc, tp, F32[E_L1W] + (long)i * DL,
                                         F32[E_L1B] + (long)i * DL, h_enc);
        launch_gemm(h_enc, F32[E_W1] + (long)i * DF, F32[E_B1] + (long)i * FL, tf, kF, kD, 1, stream);
        launch_gemm(tf, F32[E_W2] + (long)i * DF, F32[E_B2] + (long)i * DL, tp, kD, kF, 0, stream);
        k_add_ln<<<kM, 256, 0, stream>>>(h_enc, tp, F32[E_L2W] + (long)i * DL,
                                         F32[E_L2B] + (long)i * DL, h_enc);
    }

    // ---------------- decoder ----------------
    k_embed_ln<<<kM, 256, 0, stream>>>(labels, F32[I_TOK], F32[I_POS],
                                       F32[I_LNW], F32[I_LNB], h_dec);
    for (int i = 0; i < kL; ++i) {
        // self-attention
        launch_gemm(h_dec, F32[D_SWQ] + (long)i * DD, F32[D_SBQ] + (long)i * DL, tq, kD, kD, 0, stream);
        launch_gemm(h_dec, F32[D_SWK] + (long)i * DD, F32[D_SBK] + (long)i * DL, tk, kD, kD, 0, stream);
        launch_gemm(h_dec, F32[D_SWV] + (long)i * DD, F32[D_SBV] + (long)i * DL, tv, kD, kD, 0, stream);
        k_attn<<<attn_grid, 64, 0, stream>>>(tq, tk, tv, dmask, tctx);
        launch_gemm(tctx, F32[D_SWO] + (long)i * DD, F32[D_SBO] + (long)i * DL, tp, kD, kD, 0, stream);
        k_add_ln<<<kM, 256, 0, stream>>>(h_dec, tp, F32[D_L1W] + (long)i * DL,
                                         F32[D_L1B] + (long)i * DL, h_dec);
        // cross-attention (kv = encoder output)
        launch_gemm(h_dec, F32[D_CWQ] + (long)i * DD, F32[D_CBQ] + (long)i * DL, tq, kD, kD, 0, stream);
        launch_gemm(h_enc, F32[D_CWK] + (long)i * DD, F32[D_CBK] + (long)i * DL, tk, kD, kD, 0, stream);
        launch_gemm(h_enc, F32[D_CWV] + (long)i * DD, F32[D_CBV] + (long)i * DL, tv, kD, kD, 0, stream);
        k_attn<<<attn_grid, 64, 0, stream>>>(tq, tk, tv, amask, tctx);
        launch_gemm(tctx, F32[D_CWO] + (long)i * DD, F32[D_CBO] + (long)i * DL, tp, kD, kD, 0, stream);
        k_add_ln<<<kM, 256, 0, stream>>>(h_dec, tp, F32[D_L2W] + (long)i * DL,
                                         F32[D_L2B] + (long)i * DL, h_dec);
        // FFN
        launch_gemm(h_dec, F32[D_W1] + (long)i * DF, F32[D_B1] + (long)i * FL, tf, kF, kD, 1, stream);
        launch_gemm(tf, F32[D_W2] + (long)i * DF, F32[D_B2] + (long)i * DL, tp, kD, kF, 0, stream);
        k_add_ln<<<kM, 256, 0, stream>>>(h_dec, tp, F32[D_L3W] + (long)i * DL,
                                         F32[D_L3B] + (long)i * DL, h_dec);
    }

    // ---------------- logits: [4096,768] @ [768,32000] ----------------
    launch_gemm(h_dec, F32[I_OUTW], F32[I_OUTB], logits, kV, kD, 0, stream);
}
